// BasicBlock_59622736003616
// MI455X (gfx1250) — matrix operand order repaired and checked by reading, not compiled, not run on hardware
//
#include <hip/hip_runtime.h>

#define RWS    16384   // rows = B*L
#define DMODEL 1024
#define DINNER 2048
#define DSTATE 16
#define DTRANK 64
#define NXPROJ 96      // DT_RANK + 2*D_STATE
#define LSEQ   2048
#define NBATCH 8

#define KSTAGE  64
#define BPAD    8
#define BSTRIDE (KSTAGE + BPAD)   // 72 bf16 per column = 144B pitch (16B aligned, bank-friendly)

typedef __attribute__((ext_vector_type(8)))  __bf16 v8bf_t;
typedef __attribute__((ext_vector_type(16))) __bf16 v16bf_t;
typedef __attribute__((ext_vector_type(8)))  float  v8f_t;

union ABFrag { v16bf_t v; v8bf_t h[2]; };

__device__ __forceinline__ unsigned short f2bf(float f) {
    unsigned u = __float_as_uint(f);
    u += 0x7FFFu + ((u >> 16) & 1u);
    return (unsigned short)(u >> 16);
}

__device__ __forceinline__ float silu_f(float v) { return v / (1.0f + expf(-v)); }

// ---- CDNA5 async global->LDS copy (ASYNCcnt-tracked)
// builtin signature (from hipcc diagnostic): (v4i AS(1)*, v4i AS(3)*, imm offset, imm cpol)
__device__ __forceinline__ void async_b128(const unsigned short* g, unsigned short* l) {
#if __has_builtin(__builtin_amdgcn_global_load_async_to_lds_b128)
    typedef int v4i_gcc __attribute__((vector_size(16)));
    __builtin_amdgcn_global_load_async_to_lds_b128(
        (__attribute__((address_space(1))) v4i_gcc*)g,
        (__attribute__((address_space(3))) v4i_gcc*)l, 0, 0);
#else
    asm volatile("global_load_async_to_lds_b128 %0, %1, off"
                 :: "v"((unsigned)(unsigned long long)l), "v"(g)
                 : "memory");
#endif
}

__device__ __forceinline__ void wait_async_le2() {
#if __has_builtin(__builtin_amdgcn_s_wait_asynccnt)
    __builtin_amdgcn_s_wait_asynccnt(2);
#else
    asm volatile("s_wait_asynccnt 0x2" ::: "memory");
#endif
}

__device__ __forceinline__ void wait_async_le0() {
#if __has_builtin(__builtin_amdgcn_s_wait_asynccnt)
    __builtin_amdgcn_s_wait_asynccnt(0);
#else
    asm volatile("s_wait_asynccnt 0x0" ::: "memory");
#endif
}

// ---------------------------------------------------------------- f32 -> bf16
__global__ void k_cvt_bf16(const float* __restrict__ s, unsigned short* __restrict__ d, int n) {
    int i = blockIdx.x * blockDim.x + threadIdx.x;
    if (i < n) d[i] = f2bf(s[i]);
}

// ---------------------------------------------------------------- LayerNorm -> bf16
__global__ void k_ln_bf16(const float* __restrict__ x, const float* __restrict__ w,
                          const float* __restrict__ b, unsigned short* __restrict__ out) {
    const int row = blockIdx.x;
    const float* xr = x + (size_t)row * DMODEL;
    float4 v = ((const float4*)xr)[threadIdx.x];
    float s  = v.x + v.y + v.z + v.w;
    float s2 = v.x * v.x + v.y * v.y + v.z * v.z + v.w * v.w;
    #pragma unroll
    for (int off = 16; off > 0; off >>= 1) {
        s  += __shfl_xor(s,  off, 32);
        s2 += __shfl_xor(s2, off, 32);
    }
    __shared__ float ps[8], ps2[8];
    const int wid = threadIdx.x >> 5, ln = threadIdx.x & 31;
    if (ln == 0) { ps[wid] = s; ps2[wid] = s2; }
    __syncthreads();
    s = 0.f; s2 = 0.f;
    #pragma unroll
    for (int i = 0; i < 8; ++i) { s += ps[i]; s2 += ps2[i]; }
    const float mu   = s * (1.0f / DMODEL);
    const float var  = s2 * (1.0f / DMODEL) - mu * mu;
    const float rinv = rsqrtf(var + 1e-5f);
    const int base = threadIdx.x * 4;
    unsigned short r0 = f2bf((v.x - mu) * rinv * w[base + 0] + b[base + 0]);
    unsigned short r1 = f2bf((v.y - mu) * rinv * w[base + 1] + b[base + 1]);
    unsigned short r2 = f2bf((v.z - mu) * rinv * w[base + 2] + b[base + 2]);
    unsigned short r3 = f2bf((v.w - mu) * rinv * w[base + 3] + b[base + 3]);
    uint2 p; p.x = (unsigned)r0 | ((unsigned)r1 << 16); p.y = (unsigned)r2 | ((unsigned)r3 << 16);
    ((uint2*)(out + (size_t)row * DMODEL))[threadIdx.x] = p;
}

// ---------------------------------------------------------------- WMMA NT-GEMM, LDS-staged B
// C[M,N] (f32) = A[M,K](bf16) x B[N,K](bf16)^T (+ optional residual)
// block = 256 threads = 8 waves; block tile 128(M) x 64(N); wave tile 16x64.
// B strip (64 cols x 64 K) double-buffered in LDS via async global->LDS copies.
__global__ void __launch_bounds__(256)
k_gemm_bf16_nt(const unsigned short* __restrict__ Au,
               const unsigned short* __restrict__ Bu,
               float* __restrict__ C, const float* __restrict__ Res,
               int M, int N, int K) {
    __shared__ __attribute__((aligned(16))) unsigned short Bs[2][64 * BSTRIDE];

    const int tid     = threadIdx.x;
    const int lane    = tid & 31;
    const int wv      = tid >> 5;                 // 0..7 -> M sub-tile
    const int tiles_n = (N + 63) >> 6;
    const int bm      = (int)blockIdx.x / tiles_n;
    const int bn      = (int)blockIdx.x % tiles_n;
    const int col0    = bn << 6;
    const int row0    = (bm << 7) + (wv << 4);

    const int half = lane >> 4;
    const int n    = lane & 15;

    // cooperative B staging: thread covers (column bcol, quarter sub) of the 64x64 strip
    const int bcol = tid >> 2;                    // 0..63
    const int sub  = tid & 3;                     // 0..3 -> 16 bf16 each
    int cg = col0 + bcol; if (cg >= N) cg = N - 1;
    const unsigned short* gB = Bu + (size_t)cg * K + sub * 16;
    unsigned short* lB[2] = { &Bs[0][bcol * BSTRIDE + sub * 16],
                              &Bs[1][bcol * BSTRIDE + sub * 16] };

    // A fragment pointer (k order chosen here, matched by B): elems 0..15 at K=16*half+s (contiguous)
    const __bf16* pa = (const __bf16*)Au + (size_t)(row0 + n) * K + (half << 4);

    const int c0 = col0 + n, c1 = col0 + 16 + n, c2 = col0 + 32 + n, c3 = col0 + 48 + n;

    v8f_t acc0 = {}, acc1 = {}, acc2 = {}, acc3 = {};

    // prologue: stage 0
    async_b128(gB + 0, lB[0] + 0);
    async_b128(gB + 8, lB[0] + 8);

    const int nstages = K / KSTAGE;
    for (int s = 0; s < nstages; ++s) {
        if (s + 1 < nstages) {
            const unsigned short* g = gB + (size_t)(s + 1) * KSTAGE;
            unsigned short* l = lB[(s + 1) & 1];
            async_b128(g + 0, l + 0);
            async_b128(g + 8, l + 8);
            wait_async_le2();                     // stage s copies complete (in-order)
        } else {
            wait_async_le0();
        }
        __syncthreads();

        const unsigned short* bs = Bs[s & 1];
        #pragma unroll
        for (int kk = 0; kk < 2; ++kk) {          // two 32-K WMMA substeps per stage
            ABFrag a, b0, b1, b2, b3;
            a.h[0] = *(const v8bf_t*)(pa);
            a.h[1] = *(const v8bf_t*)(pa + 8);
            __builtin_prefetch(pa + 128, 0, 3);
            pa += 32;
            const int fo = kk * 32 + (half << 4); // 16 contiguous K starting at 16*half
            b0.h[0] = *(const v8bf_t*)&bs[(n     ) * BSTRIDE + fo];
            b0.h[1] = *(const v8bf_t*)&bs[(n     ) * BSTRIDE + fo + 8];
            b1.h[0] = *(const v8bf_t*)&bs[(n + 16) * BSTRIDE + fo];
            b1.h[1] = *(const v8bf_t*)&bs[(n + 16) * BSTRIDE + fo + 8];
            b2.h[0] = *(const v8bf_t*)&bs[(n + 32) * BSTRIDE + fo];
            b2.h[1] = *(const v8bf_t*)&bs[(n + 32) * BSTRIDE + fo + 8];
            b3.h[0] = *(const v8bf_t*)&bs[(n + 48) * BSTRIDE + fo];
            b3.h[1] = *(const v8bf_t*)&bs[(n + 48) * BSTRIDE + fo + 8];
            acc0 = __builtin_amdgcn_wmma_f32_16x16x32_bf16(false, a.v, false, b0.v, (short)0, acc0, false, false);
            acc1 = __builtin_amdgcn_wmma_f32_16x16x32_bf16(false, a.v, false, b1.v, (short)0, acc1, false, false);
            acc2 = __builtin_amdgcn_wmma_f32_16x16x32_bf16(false, a.v, false, b2.v, (short)0, acc2, false, false);
            acc3 = __builtin_amdgcn_wmma_f32_16x16x32_bf16(false, a.v, false, b3.v, (short)0, acc3, false, false);
        }
        __syncthreads();
    }

    // C/D layout: elem r -> row = row0 + 8*half + r, col = col0 + 16*t + n
    const int rbase = row0 + (half << 3);
    #pragma unroll
    for (int r = 0; r < 8; ++r) {
        const size_t rowoff = (size_t)(rbase + r) * N;
        if (c0 < N) { float v = acc0[r]; if (Res) v += Res[rowoff + c0]; C[rowoff + c0] = v; }
        if (c1 < N) { float v = acc1[r]; if (Res) v += Res[rowoff + c1]; C[rowoff + c1] = v; }
        if (c2 < N) { float v = acc2[r]; if (Res) v += Res[rowoff + c2]; C[rowoff + c2] = v; }
        if (c3 < N) { float v = acc3[r]; if (Res) v += Res[rowoff + c3]; C[rowoff + c3] = v; }
    }
}

// ---------------------------------------------------------------- causal depthwise conv4 + SiLU
__global__ void k_conv_silu(const float* __restrict__ xz, const float* __restrict__ cw,
                            const float* __restrict__ cb, float* __restrict__ xm,
                            unsigned short* __restrict__ xmbf) {
    size_t i = (size_t)blockIdx.x * blockDim.x + threadIdx.x;
    if (i >= (size_t)RWS * DINNER) return;
    const int    c = (int)(i & (DINNER - 1));
    const size_t r = i >> 11;                 // row = b*L + l
    const int    l = (int)(r & (LSEQ - 1));
    const float w0 = cw[c * 4 + 0], w1 = cw[c * 4 + 1], w2 = cw[c * 4 + 2], w3 = cw[c * 4 + 3];
    const float* base = xz + r * (2 * DINNER) + c;
    float acc = cb[c] + w3 * base[0];
    if (l >= 1) acc += w2 * base[-(ptrdiff_t)(2 * DINNER)];
    if (l >= 2) acc += w1 * base[-(ptrdiff_t)(4 * DINNER)];
    if (l >= 3) acc += w0 * base[-(ptrdiff_t)(6 * DINNER)];
    const float s = silu_f(acc);
    xm[i] = s;
    xmbf[i] = f2bf(s);
}

// ---------------------------------------------------------------- dt_r slice -> bf16
__global__ void k_dtr_bf16(const float* __restrict__ xdbl, unsigned short* __restrict__ dtrbf) {
    size_t i = (size_t)blockIdx.x * blockDim.x + threadIdx.x;
    if (i >= (size_t)RWS * DTRANK) return;
    const size_t row = i >> 6;
    const int    c   = (int)(i & (DTRANK - 1));
    dtrbf[i] = f2bf(xdbl[row * NXPROJ + c]);
}

// ---------------------------------------------------------------- dt = softplus(dt_pre + b)
__global__ void k_bias_softplus(float* __restrict__ dt, const float* __restrict__ b) {
    size_t i = (size_t)blockIdx.x * blockDim.x + threadIdx.x;
    if (i >= (size_t)RWS * DINNER) return;
    const int c = (int)(i & (DINNER - 1));
    const float v = dt[i] + b[c];
    dt[i] = (v > 20.0f) ? v : log1pf(expf(v));
}

// ---------------------------------------------------------------- selective scan
__global__ void k_scan(const float* __restrict__ dt, const float* __restrict__ xdbl,
                       const float* __restrict__ xm, const float* __restrict__ xz,
                       const float* __restrict__ A_log, const float* __restrict__ Dp,
                       unsigned short* __restrict__ ybf) {
    const int b = blockIdx.x >> 3;
    const int d = ((blockIdx.x & 7) << 8) + threadIdx.x;
    float A[DSTATE], st[DSTATE];
    #pragma unroll
    for (int n = 0; n < DSTATE; ++n) { A[n] = -expf(A_log[d * DSTATE + n]); st[n] = 0.f; }
    const float Dv = Dp[d];
    __shared__ float sB[DSTATE], sC[DSTATE];
    for (int l = 0; l < LSEQ; ++l) {
        const size_t row = (size_t)b * LSEQ + l;
        if (threadIdx.x < 16)      sB[threadIdx.x]      = xdbl[row * NXPROJ + DTRANK + threadIdx.x];
        else if (threadIdx.x < 32) sC[threadIdx.x - 16] = xdbl[row * NXPROJ + DTRANK + threadIdx.x];
        __syncthreads();
        const float dtv = dt[row * DINNER + d];
        const float xv  = xm[row * DINNER + d];
        const float dx  = dtv * xv;
        float y = 0.f;
        #pragma unroll
        for (int n = 0; n < DSTATE; ++n) {
            const float dA = expf(dtv * A[n]);
            st[n] = dA * st[n] + dx * sB[n];
            y += st[n] * sC[n];
        }
        const float zv = xz[row * (2 * DINNER) + DINNER + d];
        ybf[row * DINNER + d] = f2bf((y + xv * Dv) * silu_f(zv));
        __syncthreads();
    }
}

// ---------------------------------------------------------------- launcher
extern "C" void kernel_launch(void* const* d_in, const int* in_sizes, int n_in,
                              void* d_out, int out_size, void* d_ws, size_t ws_size,
                              hipStream_t stream) {
    const float* x         = (const float*)d_in[0];
    const float* ln_w      = (const float*)d_in[1];
    const float* ln_b      = (const float*)d_in[2];
    const float* in_proj_w = (const float*)d_in[3];
    const float* conv_w    = (const float*)d_in[4];
    const float* conv_b    = (const float*)d_in[5];
    const float* x_proj_w  = (const float*)d_in[6];
    const float* dt_proj_w = (const float*)d_in[7];
    const float* dt_proj_b = (const float*)d_in[8];
    const float* A_log     = (const float*)d_in[9];
    const float* Dp        = (const float*)d_in[10];
    const float* out_pw    = (const float*)d_in[11];
    float* out = (float*)d_out;

    char* ws = (char*)d_ws;
    size_t off = 0;
    auto carve = [&](size_t bytes) { char* p = ws + off; off += (bytes + 255) & ~(size_t)255; return p; };
    unsigned short* hbf    = (unsigned short*)carve((size_t)RWS * DMODEL * 2);
    unsigned short* w_in   = (unsigned short*)carve((size_t)2 * DINNER * DMODEL * 2);
    unsigned short* w_xp   = (unsigned short*)carve((size_t)NXPROJ * DINNER * 2);
    unsigned short* w_dt   = (unsigned short*)carve((size_t)DINNER * DTRANK * 2);
    unsigned short* w_out  = (unsigned short*)carve((size_t)DMODEL * DINNER * 2);
    float*          xz     = (float*)carve((size_t)RWS * 2 * DINNER * 4);
    float*          xm     = (float*)carve((size_t)RWS * DINNER * 4);
    unsigned short* xmbf   = (unsigned short*)carve((size_t)RWS * DINNER * 2);
    float*          xdbl   = (float*)carve((size_t)RWS * NXPROJ * 4);
    unsigned short* dtrbf  = (unsigned short*)carve((size_t)RWS * DTRANK * 2);
    float*          dtb    = (float*)carve((size_t)RWS * DINNER * 4);
    unsigned short* ybf    = (unsigned short*)carve((size_t)RWS * DINNER * 2);

    const int TPB = 256;
    auto cvt = [&](const float* s, unsigned short* d, int n) {
        k_cvt_bf16<<<(n + TPB - 1) / TPB, TPB, 0, stream>>>(s, d, n);
    };
    cvt(in_proj_w, w_in,  2 * DINNER * DMODEL);
    cvt(x_proj_w,  w_xp,  NXPROJ * DINNER);
    cvt(dt_proj_w, w_dt,  DINNER * DTRANK);
    cvt(out_pw,    w_out, DMODEL * DINNER);

    k_ln_bf16<<<RWS, TPB, 0, stream>>>(x, ln_w, ln_b, hbf);

    auto gemm = [&](const unsigned short* A, const unsigned short* B, float* C,
                    const float* Res, int M, int N, int K) {
        const int blocks = (M >> 7) * ((N + 63) >> 6);   // 128x64 block tiles
        k_gemm_bf16_nt<<<blocks, TPB, 0, stream>>>(A, B, C, Res, M, N, K);
    };

    // xz = h @ in_proj_w^T
    gemm(hbf, w_in, xz, nullptr, RWS, 2 * DINNER, DMODEL);

    const size_t nact = (size_t)RWS * DINNER;
    k_conv_silu<<<(unsigned)((nact + TPB - 1) / TPB), TPB, 0, stream>>>(xz, conv_w, conv_b, xm, xmbf);

    // x_dbl = xm @ x_proj_w^T
    gemm(xmbf, w_xp, xdbl, nullptr, RWS, NXPROJ, DINNER);

    k_dtr_bf16<<<(RWS * DTRANK) / TPB, TPB, 0, stream>>>(xdbl, dtrbf);

    // dt_pre = dt_r @ dt_proj_w^T ; then softplus(+bias)
    gemm(dtrbf, w_dt, dtb, nullptr, RWS, DINNER, DTRANK);
    k_bias_softplus<<<(unsigned)((nact + TPB - 1) / TPB), TPB, 0, stream>>>(dtb, dt_proj_b);

    // selective scan (fused D-skip + silu(z) gating), bf16 output
    k_scan<<<NBATCH * (DINNER / TPB), TPB, 0, stream>>>(dtb, xdbl, xm, xz, A_log, Dp, ybf);

    // out = y @ out_proj_w^T + x (fused residual)
    gemm(ybf, w_out, out, x, RWS, DMODEL, DINNER);
}
